// MemNet_12773232738429
// MI455X (gfx1250) — matrix operand order repaired and checked by reading, not compiled, not run on hardware
//
#include <hip/hip_runtime.h>
#include <math.h>

// ---------------------------------------------------------------------------
// MemNet forward for MI455X (gfx1250, wave32, WMMA).
//
// Decomposition:
//   prep kernels : bf16/packed weight conversion, W_out transpose, embedding
//                  gather, memory init
//   gemm #1      : gates_x = emb_seq @ W_ih[:, :256].T   (WMMA bf16)
//   recurrence   : 16 blocks (1 per batch elem), serial over T=128; LSTM +
//                  top-k content addressing; writes h_seq (bf16)
//   gemm #2      : logits = h_seq @ W_out + b_out        (WMMA bf16, 67 GFLOP)
// ---------------------------------------------------------------------------

typedef __bf16 bf16_t;
typedef __attribute__((ext_vector_type(16))) __bf16 v16bf;
typedef __attribute__((ext_vector_type(8)))  __bf16 v8bf;
typedef __attribute__((ext_vector_type(8)))  float  v8f;

#define VOCAB   32000
#define EMBED   256
#define HIDDEN  512
#define SLOTS   1024
#define MEM_D   64
#define HEADS   4
#define TOPK    8
#define BATCH   16
#define TSTEPS  128
#define ROWS    (BATCH*TSTEPS)      // 2048
#define GATES   (4*HIDDEN)          // 2048
#define KSER    (MEM_D + HIDDEN)    // 576 serial K per gate
#define BETA_C  (1.0f + 1e-6f)
#define EPS_C   1e-8f

__device__ __forceinline__ float bfu_lo(unsigned int u){ return __uint_as_float(u << 16); }
__device__ __forceinline__ float bfu_hi(unsigned int u){ return __uint_as_float(u & 0xffff0000u); }
__device__ __forceinline__ unsigned int pack2(float lo, float hi){
    unsigned short l = __builtin_bit_cast(unsigned short, (bf16_t)lo);
    unsigned short h = __builtin_bit_cast(unsigned short, (bf16_t)hi);
    return (unsigned int)l | ((unsigned int)h << 16);
}
__device__ __forceinline__ float sigf(float x){ return 1.0f / (1.0f + expf(-x)); }

// ======================= WMMA GEMM: C[M,N] = A[M,K] * Bt[N,K]^T (+bias) =====
// block tile 64(M) x 128(N), 256 thr = 8 waves (2 Mwaves x 4 Nwaves),
// wave tile 32x32 = 2x2 of 16x16x32 bf16 WMMA tiles, K step 32, LDS staged.
__global__ __launch_bounds__(256)
void wmma_gemm_bf16(const bf16_t* __restrict__ A, const bf16_t* __restrict__ Bt,
                    float* __restrict__ C, const float* __restrict__ bias,
                    int M, int N, int K)
{
    __shared__ __align__(16) bf16_t As[64][40];    // pad: 40*2=80B rows (16B aligned)
    __shared__ __align__(16) bf16_t Bs[128][40];
    const int tid  = threadIdx.x;
    const int m0   = blockIdx.y * 64;
    const int n0   = blockIdx.x * 128;
    const int wave = tid >> 5, lane = tid & 31;
    const int half = lane >> 4, l16 = lane & 15;
    const int wm   = (wave & 1) * 32;
    const int wn   = (wave >> 1) * 32;

    v8f acc[2][2] = {};

    const int ar = tid >> 2, ac = (tid & 3) * 8;    // A tile 64x32
    const int br = tid >> 1, bc = (tid & 1) * 16;   // Bt tile 128x32

    for (int kk = 0; kk < K; kk += 32) {
        __syncthreads();
        *(uint4*)&As[ar][ac]   = *(const uint4*)&A [(size_t)(m0+ar)*K + kk + ac];
        *(uint4*)&Bs[br][bc]   = *(const uint4*)&Bt[(size_t)(n0+br)*K + kk + bc];
        *(uint4*)&Bs[br][bc+8] = *(const uint4*)&Bt[(size_t)(n0+br)*K + kk + bc + 8];
        __syncthreads();

        v16bf afrag[2], bfrag[2];
#pragma unroll
        for (int i = 0; i < 2; i++) {
            // A frag: lanes<16 K={e,16+e'}, lanes>=16 K={8+e,24+e'} -> lo at half*8, hi at +16
            const bf16_t* pa = &As[wm + 16*i + l16][half*8];
            v8bf alo = *(const v8bf*)pa;
            v8bf ahi = *(const v8bf*)(pa + 16);
            afrag[i] = __builtin_shufflevector(alo, ahi, 0,1,2,3,4,5,6,7,8,9,10,11,12,13,14,15);
            // B frag: same K order as A -> lo at half*8, hi at +16
            const bf16_t* pb = &Bs[wn + 16*i + l16][half*8];
            v8bf blo = *(const v8bf*)pb;
            v8bf bhi = *(const v8bf*)(pb + 16);
            bfrag[i] = __builtin_shufflevector(blo, bhi, 0,1,2,3,4,5,6,7,8,9,10,11,12,13,14,15);
        }
#pragma unroll
        for (int i = 0; i < 2; i++)
#pragma unroll
            for (int j = 0; j < 2; j++)
                acc[i][j] = __builtin_amdgcn_wmma_f32_16x16x32_bf16(
                    false, afrag[i], false, bfrag[j], (short)0, acc[i][j], false, false);
    }

#pragma unroll
    for (int i = 0; i < 2; i++)
#pragma unroll
        for (int j = 0; j < 2; j++) {
            int gn = n0 + wn + 16*j + l16;
            float bv = bias ? bias[gn] : 0.0f;
#pragma unroll
            for (int r = 0; r < 8; r++) {
                int gm = m0 + wm + 16*i + r + half*8;   // C/D layout: M = r + 8*half
                C[(size_t)gm * N + gn] = acc[i][j][r] + bv;
            }
        }
}

// ============================ prep kernels ==================================
__global__ void k_transpose_wout(const float* __restrict__ W, bf16_t* __restrict__ Wt)
{   // W [512][32000] f32 -> Wt [32000][512] bf16 via LDS tile
    __shared__ float tile[32][33];
    int bn = blockIdx.x * 32, bk = blockIdx.y * 32;
    int tx = threadIdx.x & 31, ty = threadIdx.x >> 5;   // 32 x 8
    for (int i = 0; i < 32; i += 8)
        tile[ty+i][tx] = W[(size_t)(bk+ty+i)*VOCAB + bn + tx];
    __syncthreads();
    for (int i = 0; i < 32; i += 8)
        Wt[(size_t)(bn+ty+i)*HIDDEN + bk + tx] = (bf16_t)tile[tx][ty+i];
}

__global__ void k_wihx(const float* __restrict__ W_ih, bf16_t* __restrict__ out)
{   // out [2048 j][256 k] = bf16(W_ih[j][k])   (Bt layout for gemm #1)
    int idx = blockIdx.x*256 + threadIdx.x;
    int j = idx >> 8, k = idx & 255;
    out[idx] = (bf16_t)W_ih[(size_t)j*(EMBED+MEM_D) + k];
}

__global__ void k_pack_whh(const float* __restrict__ W_hh, unsigned int* __restrict__ out)
{   // out [256 m][2048 j] = pack(W_hh[j][2m], W_hh[j][2m+1])
    int idx = blockIdx.x*256 + threadIdx.x;
    int m = idx >> 11, j = idx & 2047;
    out[idx] = pack2(W_hh[(size_t)j*HIDDEN + 2*m], W_hh[(size_t)j*HIDDEN + 2*m + 1]);
}

__global__ void k_pack_wihr(const float* __restrict__ W_ih, unsigned int* __restrict__ out)
{   // out [32 m][2048 j] = pack(W_ih[j][256+2m], W_ih[j][256+2m+1])
    int idx = blockIdx.x*256 + threadIdx.x;
    int m = idx >> 11, j = idx & 2047;
    out[idx] = pack2(W_ih[(size_t)j*(EMBED+MEM_D) + EMBED + 2*m],
                     W_ih[(size_t)j*(EMBED+MEM_D) + EMBED + 2*m + 1]);
}

__global__ void k_pack_heads(const float* __restrict__ Wrk, const float* __restrict__ Wwk,
                             const float* __restrict__ Wwv, const float* __restrict__ Wer,
                             unsigned int* __restrict__ out)
{   // out [256 m][1024 c], c = mat*256+col; pack over K pairs
    int idx = blockIdx.x*256 + threadIdx.x;
    int m = idx >> 10, c = idx & 1023;
    int mat = c >> 8, col = c & 255;
    const float* W = (mat==0) ? Wrk : (mat==1) ? Wwk : (mat==2) ? Wwv : Wer;
    out[idx] = pack2(W[(size_t)(2*m)*256 + col], W[(size_t)(2*m+1)*256 + col]);
}

__global__ void k_biases(const float* __restrict__ b_ih, const float* __restrict__ b_hh,
                         const float* __restrict__ b_rk, const float* __restrict__ b_wk,
                         const float* __restrict__ b_wv, const float* __restrict__ b_er,
                         float* __restrict__ bias_comb, float* __restrict__ b_heads)
{
    int idx = blockIdx.x*256 + threadIdx.x;
    if (idx < GATES) bias_comb[idx] = b_ih[idx] + b_hh[idx];
    else if (idx < GATES + 1024) {
        int c = idx - GATES, mat = c >> 8, col = c & 255;
        const float* bb = (mat==0) ? b_rk : (mat==1) ? b_wk : (mat==2) ? b_wv : b_er;
        b_heads[c] = bb[col];
    }
}

__global__ void k_embed(const int* __restrict__ seq, const float* __restrict__ E,
                        bf16_t* __restrict__ out)
{   // out [2048 r][256 d] = bf16(E[token(r)][d]);  r = b*T + t matches input_seq [B,T]
    int idx = blockIdx.x*256 + threadIdx.x;
    int r = idx >> 8, d = idx & 255;
    int tok = seq[r];
    out[idx] = (bf16_t)E[(size_t)tok*EMBED + d];
}

__global__ void k_meminit(float* __restrict__ memT)
{
    int idx = blockIdx.x*256 + threadIdx.x;   // 16*64*1024 = 1M
    memT[idx] = 1e-6f;
}

// ======================= recurrence (1 block per batch elem) ================
__device__ __forceinline__ void do_address(
    int tid, const float keys[HEADS][MEM_D], const float* __restrict__ memTb,
    const float* sh_norms, float (*sh_scores)[SLOTS],
    int (*sh_topi)[TOPK], float (*sh_w)[TOPK], float* sh_kscale)
{
    if (tid < HEADS) {
        float ss = 0.0f;
        for (int d = 0; d < MEM_D; d++) { float v = keys[tid][d]; ss += v*v; }
        sh_kscale[tid] = BETA_C / (sqrtf(ss) + EPS_C);
    }
    __syncthreads();
    {   // scores: one slot per thread, 4 heads per memory read
        int s = tid;
        float a0=0,a1=0,a2=0,a3=0;
        for (int d = 0; d < MEM_D; d++) {
            float m = memTb[d*SLOTS + s];
            a0 += keys[0][d]*m; a1 += keys[1][d]*m;
            a2 += keys[2][d]*m; a3 += keys[3][d]*m;
        }
        float inv = 1.0f / sh_norms[s];
        sh_scores[0][s] = a0 * sh_kscale[0] * inv;
        sh_scores[1][s] = a1 * sh_kscale[1] * inv;
        sh_scores[2][s] = a2 * sh_kscale[2] * inv;
        sh_scores[3][s] = a3 * sh_kscale[3] * inv;
    }
    __syncthreads();
    // top-8 + softmax: wave h handles head h, pure wave32 shuffles (no barriers)
    int wave = tid >> 5, lane = tid & 31;
    if (wave < HEADS) {
        int h = wave;
        float vals[TOPK];
        for (int kk = 0; kk < TOPK; kk++) {
            float bv = -3.0e38f; int bi = 0;
            for (int s = lane; s < SLOTS; s += 32) {
                float v = sh_scores[h][s];
                if (v > bv) { bv = v; bi = s; }
            }
            for (int off = 16; off; off >>= 1) {
                float ov = __shfl_xor(bv, off, 32);
                int   oi = __shfl_xor(bi, off, 32);
                if (ov > bv || (ov == bv && oi < bi)) { bv = ov; bi = oi; }
            }
            vals[kk] = bv;
            if (lane == 0) { sh_topi[h][kk] = bi; sh_scores[h][bi] = -3.0e38f; }
        }
        if (lane == 0) {
            float mx = vals[0], s = 0.0f, e[TOPK];
            for (int kk = 0; kk < TOPK; kk++) { e[kk] = expf(vals[kk]-mx); s += e[kk]; }
            float is = 1.0f / s;
            for (int kk = 0; kk < TOPK; kk++) sh_w[h][kk] = e[kk]*is;
        }
    }
    __syncthreads();
}

__global__ __launch_bounds__(1024)
void recurrence_kernel(const float* __restrict__ gates_x,
                       const unsigned int* __restrict__ whh_pk,
                       const unsigned int* __restrict__ wihr_pk,
                       const float* __restrict__ bias_comb,
                       const unsigned int* __restrict__ hd_pk,
                       const float* __restrict__ b_heads,
                       float* __restrict__ memT,
                       bf16_t* __restrict__ h_seq)
{
    __shared__ float sh_h[HIDDEN], sh_c[HIDDEN], sh_read[MEM_D];
    __shared__ float sh_gates[GATES];
    __shared__ float sh_keys[4][HEADS][MEM_D];   // 0 rk, 1 wk, 2 wv, 3 er
    __shared__ float sh_norms[SLOTS];
    __shared__ float sh_scores[HEADS][SLOTS];
    __shared__ int   sh_topi[HEADS][TOPK];
    __shared__ float sh_w[HEADS][TOPK];
    __shared__ float sh_kscale[HEADS];

    const int tid = threadIdx.x;
    const int b   = blockIdx.x;
    float* memTb = memT + (size_t)b * MEM_D * SLOTS;   // [d][s]

    if (tid < HIDDEN) { sh_h[tid] = 0.0f; sh_c[tid] = 0.0f; }
    if (tid < MEM_D)  sh_read[tid] = 0.0f;
    __syncthreads();

    for (int t = 0; t < TSTEPS; t++) {
        const int r = b * TSTEPS + t;

        // ---- gates = gates_x (precomputed emb part) + read@WihR + h@Whh ----
#pragma unroll
        for (int jj = tid; jj < GATES; jj += 1024) {
            float acc = gates_x[(size_t)r*GATES + jj] + bias_comb[jj];
            if (t + 1 < TSTEPS)
                __builtin_prefetch(&gates_x[(size_t)(r+1)*GATES + jj], 0, 1);
#pragma unroll 8
            for (int m = 0; m < MEM_D/2; m++) {
                unsigned int u = wihr_pk[m*GATES + jj];
                acc += sh_read[2*m]*bfu_lo(u) + sh_read[2*m+1]*bfu_hi(u);
            }
#pragma unroll 8
            for (int m = 0; m < HIDDEN/2; m++) {
                unsigned int u = whh_pk[m*GATES + jj];
                acc += sh_h[2*m]*bfu_lo(u) + sh_h[2*m+1]*bfu_hi(u);
            }
            sh_gates[jj] = acc;
        }
        __syncthreads();

        // ---- LSTM cell update (torch gate order i,f,g,o) ----
        if (tid < HIDDEN) {
            float ig = sigf(sh_gates[tid]);
            float fg = sigf(sh_gates[tid + HIDDEN]);
            float gg = tanhf(sh_gates[tid + 2*HIDDEN]);
            float og = sigf(sh_gates[tid + 3*HIDDEN]);
            float c  = fg * sh_c[tid] + ig * gg;
            float h  = og * tanhf(c);
            sh_c[tid] = c; sh_h[tid] = h;
            h_seq[(size_t)r*HIDDEN + tid] = (bf16_t)h;
        }
        __syncthreads();

        // ---- head projections: 1024 outputs (4 mats x 256), h@W ----
        {
            float acc = b_heads[tid];
#pragma unroll 8
            for (int m = 0; m < HIDDEN/2; m++) {
                unsigned int u = hd_pk[m*1024 + tid];
                acc += sh_h[2*m]*bfu_lo(u) + sh_h[2*m+1]*bfu_hi(u);
            }
            int mat = tid >> 8, hh = (tid >> 6) & 3, dd = tid & 63;
            if (mat == 3) acc = sigf(acc);
            sh_keys[mat][hh][dd] = acc;
        }
        __syncthreads();

        // ---- memory slot norms (pre-write memory; shared by both addressings)
        {
            int s = tid;
            float ss = 0.0f;
            for (int d = 0; d < MEM_D; d++) { float m = memTb[d*SLOTS + s]; ss += m*m; }
            sh_norms[s] = sqrtf(ss) + EPS_C;
        }
        __syncthreads();

        // ---- read addressing + read vector ----
        do_address(tid, sh_keys[0], memTb, sh_norms, sh_scores, sh_topi, sh_w, sh_kscale);
        if (tid < MEM_D) {
            float acc = 0.0f;
            for (int h = 0; h < HEADS; h++)
                for (int k = 0; k < TOPK; k++)
                    acc += sh_w[h][k] * memTb[tid*SLOTS + sh_topi[h][k]];
            sh_read[tid] = acc * (1.0f / HEADS);
        }
        __syncthreads();

        // ---- write addressing + sparse erase/add update ----
        do_address(tid, sh_keys[1], memTb, sh_norms, sh_scores, sh_topi, sh_w, sh_kscale);
        {
            int s = tid;
            float wm[HEADS]; bool any = false;
            for (int h = 0; h < HEADS; h++) {
                float w = 0.0f;
                for (int k = 0; k < TOPK; k++) if (sh_topi[h][k] == s) w = sh_w[h][k];
                wm[h] = w; any |= (w != 0.0f);
            }
            if (any) {
                for (int d = 0; d < MEM_D; d++) {
                    float keep = 1.0f, add = 0.0f;
                    for (int h = 0; h < HEADS; h++) {
                        keep *= (1.0f - wm[h] * sh_keys[3][h][d]);
                        add  += wm[h] * sh_keys[2][h][d];
                    }
                    float* p = memTb + d*SLOTS + s;
                    *p = (*p) * keep + add;
                }
            }
        }
        __syncthreads();
    }
}

// ============================== launcher ====================================
extern "C" void kernel_launch(void* const* d_in, const int* in_sizes, int n_in,
                              void* d_out, int out_size, void* d_ws, size_t ws_size,
                              hipStream_t stream)
{
    (void)in_sizes; (void)n_in; (void)out_size; (void)ws_size;
    const int*   input_seq = (const int*)  d_in[0];
    const float* E      = (const float*)d_in[1];
    const float* W_ih   = (const float*)d_in[2];
    const float* W_hh   = (const float*)d_in[3];
    const float* b_ih   = (const float*)d_in[4];
    const float* b_hh   = (const float*)d_in[5];
    const float* W_out  = (const float*)d_in[6];
    const float* b_out  = (const float*)d_in[7];
    const float* W_rk   = (const float*)d_in[8];
    const float* b_rk   = (const float*)d_in[9];
    const float* W_wk   = (const float*)d_in[10];
    const float* b_wk   = (const float*)d_in[11];
    const float* W_wv   = (const float*)d_in[12];
    const float* b_wv   = (const float*)d_in[13];
    const float* W_er   = (const float*)d_in[14];
    const float* b_er   = (const float*)d_in[15];

    char* ws = (char*)d_ws;
    size_t off = 0;
    auto alloc = [&](size_t bytes) -> char* {
        char* p = ws + off; off += (bytes + 255) & ~(size_t)255; return p;
    };
    bf16_t*       WoutT     = (bf16_t*)      alloc((size_t)VOCAB*HIDDEN*2);   // [N][K]
    bf16_t*       WihX      = (bf16_t*)      alloc((size_t)GATES*EMBED*2);    // [N][K]
    unsigned int* whh_pk    = (unsigned int*)alloc((size_t)(HIDDEN/2)*GATES*4);
    unsigned int* wihr_pk   = (unsigned int*)alloc((size_t)(MEM_D/2)*GATES*4);
    unsigned int* hd_pk     = (unsigned int*)alloc((size_t)(HIDDEN/2)*1024*4);
    float*        bias_comb = (float*)       alloc((size_t)GATES*4);
    float*        b_heads   = (float*)       alloc((size_t)1024*4);
    bf16_t*       emb_seq   = (bf16_t*)      alloc((size_t)ROWS*EMBED*2);
    float*        gates_x   = (float*)       alloc((size_t)ROWS*GATES*4);
    bf16_t*       h_seq     = (bf16_t*)      alloc((size_t)ROWS*HIDDEN*2);
    float*        memT      = (float*)       alloc((size_t)BATCH*MEM_D*SLOTS*4);

    // ---- prep ----
    k_transpose_wout<<<dim3(VOCAB/32, HIDDEN/32), 256, 0, stream>>>(W_out, WoutT);
    k_wihx     <<<(GATES*EMBED)/256,      256, 0, stream>>>(W_ih, WihX);
    k_pack_whh <<<((HIDDEN/2)*GATES)/256, 256, 0, stream>>>(W_hh, whh_pk);
    k_pack_wihr<<<((MEM_D/2)*GATES)/256,  256, 0, stream>>>(W_ih, wihr_pk);
    k_pack_heads<<<((HIDDEN/2)*1024)/256, 256, 0, stream>>>(W_rk, W_wk, W_wv, W_er, hd_pk);
    k_biases   <<<(GATES+1024+255)/256,   256, 0, stream>>>(b_ih, b_hh, b_rk, b_wk, b_wv, b_er,
                                                            bias_comb, b_heads);
    k_embed    <<<(ROWS*EMBED)/256,       256, 0, stream>>>(input_seq, E, emb_seq);
    k_meminit  <<<(BATCH*MEM_D*SLOTS)/256,256, 0, stream>>>(memT);

    // ---- gemm #1: emb contribution to gates (WMMA bf16) ----
    wmma_gemm_bf16<<<dim3(GATES/128, ROWS/64), 256, 0, stream>>>(
        emb_seq, WihX, gates_x, nullptr, ROWS, GATES, EMBED);

    // ---- serial recurrence: 16 independent workgroups (1 per batch elem) ---
    recurrence_kernel<<<BATCH, 1024, 0, stream>>>(
        gates_x, whh_pk, wihr_pk, bias_comb, hd_pk, b_heads, memT, h_seq);

    // ---- gemm #2: logits (WMMA bf16, 67 GFLOP, HBM-bound on 262MB write) ---
    wmma_gemm_bf16<<<dim3(VOCAB/128, ROWS/64), 256, 0, stream>>>(
        h_seq, WoutT, (float*)d_out, b_out, ROWS, VOCAB, HIDDEN);
}
